// BitLinear_3934190043739
// MI455X (gfx1250) — 8-bit matrix operand order repaired and checked by reading, compiled, not run on hardware
//
#include <hip/hip_runtime.h>
#include <stdint.h>

typedef int v8i __attribute__((ext_vector_type(8)));

static constexpr int   DIM_K   = 4096;
static constexpr int   DIM_N   = 11008;
static constexpr int   DIM_M   = 8192;
static constexpr float EPS     = 1e-5f;
static constexpr size_t W_ELEMS = (size_t)DIM_N * DIM_K;
static constexpr size_t W_VEC4  = W_ELEMS / 4;

__global__ __launch_bounds__(256) void wabs_partial_kernel(
    const float* __restrict__ w, float* __restrict__ partials)
{
    float s = 0.f;
    const float4* w4 = (const float4*)w;
    for (size_t i = (size_t)blockIdx.x * 256 + threadIdx.x; i < W_VEC4;
         i += (size_t)gridDim.x * 256) {
        float4 v = w4[i];
        s += fabsf(v.x) + fabsf(v.y) + fabsf(v.z) + fabsf(v.w);
    }
    __shared__ float red[256];
    red[threadIdx.x] = s;
    __syncthreads();
    for (int k = 128; k > 0; k >>= 1) {
        if (threadIdx.x < (unsigned)k) red[threadIdx.x] += red[threadIdx.x + k];
        __syncthreads();
    }
    if (threadIdx.x == 0) partials[blockIdx.x] = red[0];
}

__global__ __launch_bounds__(256) void wabs_final_kernel(
    const float* __restrict__ partials, float* __restrict__ rswp)
{
    float s = 0.f;
    for (int i = threadIdx.x; i < 1024; i += 256) s += partials[i];
    __shared__ float red[256];
    red[threadIdx.x] = s;
    __syncthreads();
    for (int k = 128; k > 0; k >>= 1) {
        if (threadIdx.x < (unsigned)k) red[threadIdx.x] += red[threadIdx.x + k];
        __syncthreads();
    }
    if (threadIdx.x == 0) *rswp = fmaxf(red[0] / (float)W_ELEMS, EPS);
}

__global__ __launch_bounds__(256) void wquant_kernel(
    const float* __restrict__ w, const float* __restrict__ rswp,
    int8_t* __restrict__ qw)
{
    const float ws = 1.f / *rswp;
    const size_t i = (size_t)blockIdx.x * 256 + threadIdx.x;
    float4 v = ((const float4*)w)[i];
    char4 q;
    q.x = (char)fminf(fmaxf(rintf(v.x * ws), -1.f), 1.f);
    q.y = (char)fminf(fmaxf(rintf(v.y * ws), -1.f), 1.f);
    q.z = (char)fminf(fmaxf(rintf(v.z * ws), -1.f), 1.f);
    q.w = (char)fminf(fmaxf(rintf(v.w * ws), -1.f), 1.f);
    ((char4*)qw)[i] = q;
}

__global__ __launch_bounds__(256) void actquant_kernel(
    const float* __restrict__ x, int8_t* __restrict__ qx,
    float* __restrict__ rsx)
{
    const int t = blockIdx.x;
    const float4* xr = (const float4*)(x + (size_t)t * DIM_K);
    float amax = 0.f;
    for (int i = threadIdx.x; i < DIM_K / 4; i += 256) {
        float4 v = xr[i];
        amax = fmaxf(amax, fmaxf(fmaxf(fabsf(v.x), fabsf(v.y)),
                                 fmaxf(fabsf(v.z), fabsf(v.w))));
    }
    __shared__ float red[256];
    red[threadIdx.x] = amax;
    __syncthreads();
    for (int k = 128; k > 0; k >>= 1) {
        if (threadIdx.x < (unsigned)k)
            red[threadIdx.x] = fmaxf(red[threadIdx.x], red[threadIdx.x + k]);
        __syncthreads();
    }
    const float clipped = fmaxf(red[0], EPS);
    const float scale   = 127.f / clipped;
    if (threadIdx.x == 0) rsx[t] = clipped / 127.f;
    char4* qr = (char4*)(qx + (size_t)t * DIM_K);
    for (int i = threadIdx.x; i < DIM_K / 4; i += 256) {
        float4 v = xr[i];
        char4 q;
        q.x = (char)fminf(fmaxf(rintf(v.x * scale), -128.f), 127.f);
        q.y = (char)fminf(fmaxf(rintf(v.y * scale), -128.f), 127.f);
        q.z = (char)fminf(fmaxf(rintf(v.z * scale), -128.f), 127.f);
        q.w = (char)fminf(fmaxf(rintf(v.w * scale), -128.f), 127.f);
        qr[i] = q;
    }
}

__global__ __launch_bounds__(256) void gemm_iu8_kernel(
    const int8_t* __restrict__ qx, const int8_t* __restrict__ qw,
    const float* __restrict__ rsx, const float* __restrict__ rswp,
    const float* __restrict__ bias, float* __restrict__ out)
{
    const int lane  = threadIdx.x & 31;
    const int wave  = threadIdx.x >> 5;
    const int waveN = blockIdx.x * 4 + (wave & 3);
    const int waveM = blockIdx.y * 2 + (wave >> 2);
    const int l16   = lane & 15;
    const bool hi   = lane >= 16;

    const int mBase = waveM * 64;
    const int nBase = waveN * 64;

    const int8_t* aptr = qx + (size_t)(mBase + l16) * DIM_K + (hi ? 8 : 0);
    const int8_t* bptr = qw + (size_t)(nBase + l16) * DIM_K + (hi ? 8 : 0);

    v8i acc[4][4];
    const v8i vzero = {0, 0, 0, 0, 0, 0, 0, 0};
#pragma unroll
    for (int i = 0; i < 4; ++i)
#pragma unroll
        for (int j = 0; j < 4; ++j) acc[i][j] = vzero;

    for (int k0 = 0; k0 < DIM_K; k0 += 64) {
        v8i a[4];
#pragma unroll
        for (int i = 0; i < 4; ++i) {
            const int8_t* p = aptr + (size_t)i * 16 * DIM_K + k0;
            int2 c0 = *(const int2*)(p);
            int2 c1 = *(const int2*)(p + 16);
            int2 c2 = *(const int2*)(p + 32);
            int2 c3 = *(const int2*)(p + 48);
            a[i][0] = c0.x; a[i][1] = c0.y;
            a[i][2] = c1.x; a[i][3] = c1.y;
            a[i][4] = c2.x; a[i][5] = c2.y;
            a[i][6] = c3.x; a[i][7] = c3.y;
        }
#pragma unroll
        for (int j = 0; j < 4; ++j) {
            const int8_t* p = bptr + (size_t)j * 16 * DIM_K + k0;
            int2 c0 = *(const int2*)(p);
            int2 c1 = *(const int2*)(p + 16);
            int2 c2 = *(const int2*)(p + 32);
            int2 c3 = *(const int2*)(p + 48);
            v8i b;
            b[0] = c0.x; b[1] = c0.y; b[2] = c1.x; b[3] = c1.y;
            b[4] = c2.x; b[5] = c2.y; b[6] = c3.x; b[7] = c3.y;
#pragma unroll
            for (int i = 0; i < 4; ++i)
                acc[i][j] = __builtin_amdgcn_wmma_i32_16x16x64_iu8(
                     true, a[i],  true, b, acc[i][j],
                     false,  false);
        }
    }

    const float rsw = *rswp;
#pragma unroll
    for (int j = 0; j < 4; ++j) {
        const int n = nBase + j * 16 + l16;
        const float bv = bias[n];
#pragma unroll
        for (int i = 0; i < 4; ++i) {
            const int mrow = mBase + i * 16 + (hi ? 8 : 0);
#pragma unroll
            for (int v = 0; v < 8; ++v) {
                const int m = mrow + v;
                const float s = rsx[m] * rsw;
                out[(size_t)m * DIM_N + n] = fmaf((float)acc[i][j][v], s, bv);
            }
        }
    }
}

extern "C" void kernel_launch(void* const* d_in, const int* in_sizes, int n_in,
                              void* d_out, int out_size, void* d_ws, size_t ws_size,
                              hipStream_t stream)
{
    const float* x    = (const float*)d_in[0];
    const float* w    = (const float*)d_in[1];
    const float* bias = (const float*)d_in[2];
    float* out        = (float*)d_out;

    char* ws = (char*)d_ws;
    float*  partials = (float*)ws;
    float*  rswp     = (float*)(ws + 4096);
    float*  rsx      = (float*)(ws + 8192);
    int8_t* qx       = (int8_t*)(ws + 40960);
    int8_t* qw       = (int8_t*)(ws + 40960 + (size_t)DIM_M * DIM_K);

    wabs_partial_kernel<<<1024, 256, 0, stream>>>(w, partials);
    wabs_final_kernel<<<1, 256, 0, stream>>>(partials, rswp);
    wquant_kernel<<<(unsigned)(W_VEC4 / 256), 256, 0, stream>>>(w, rswp, qw);
    actquant_kernel<<<DIM_M, 256, 0, stream>>>(x, qx, rsx);
    gemm_iu8_kernel<<<dim3(43, 64), 256, 0, stream>>>(qx, qw, rsx, rswp, bias, out);
}
